// test_net_49589692399704
// MI455X (gfx1250) — matrix operand order repaired and checked by reading, not compiled, not run on hardware
//
#include <hip/hip_runtime.h>
#include <hip/hip_bf16.h>
#include <math.h>

// Problem constants
#define NN    16384
#define A0c   64
#define A1c   32
#define XW    160      // A0 + 3*A1
#define H0c   128
#define H1c   64
#define EPSf  1e-6f

// Fused-GEMM K extents
#define KSO   5120     // 64*64 + 32*32   (open, scalar out)
#define KVO   4096     // 64*32 + 32*64   (open, vector out)
#define KSC   10240    // 64*128 + 32*64  (close, scalar out)
#define KVC   8192     // 64*64 + 32*128  (close, vector out)

typedef __bf16 bf16_t;
typedef __attribute__((ext_vector_type(16))) __bf16 v16bf;
typedef __attribute__((ext_vector_type(8)))  __bf16 v8bf;
typedef __attribute__((ext_vector_type(8)))  float  v8f;

// Workspace layout (bytes)
#define OFF_BSO 0
#define OFF_BVO (OFF_BSO + KSO * H0c * 2)
#define OFF_BSC (OFF_BVO + KVO * H1c * 2)
#define OFF_BVC (OFF_BSC + KSC * A0c * 2)
#define OFF_BMS (OFF_BVC + KVC * A1c * 2)
#define OFF_BMV (OFF_BMS + 3 * H0c * H0c * 2)
#define OFF_SA  (OFF_BMV + 3 * H1c * H1c * 2)
#define OFF_SB  (OFF_SA + NN * H0c * 4)
#define OFF_VA  (OFF_SB + NN * H0c * 4)
#define OFF_VB  (OFF_VA + NN * 3 * H1c * 4)
#define WS_NEED ((size_t)(OFF_VB + NN * 3 * H1c * 4))

__device__ inline unsigned bfbits(float f) {
  unsigned u = __builtin_bit_cast(unsigned, f);
  u += 0x7FFFu + ((u >> 16) & 1u);              // round-to-nearest-even
  return u >> 16;
}

__device__ inline bf16_t f2bf(float f) {
  unsigned short h = (unsigned short)bfbits(f);
  return __builtin_bit_cast(bf16_t, h);
}

// Pack two consecutive K elements and store with one 32-bit LDS store.
__device__ inline void store_pair(bf16_t* p, float a, float b) {
  *(unsigned*)p = bfbits(a) | (bfbits(b) << 16);
}

__device__ inline v8f zero8() {
  v8f z;
  for (int i = 0; i < 8; ++i) z[i] = 0.f;
  return z;
}

__device__ inline v8f wmma_bf16(v16bf a, v16bf b, v8f c) {
  // 8 args: (neg_a, A, neg_b, B, c_mod, C, reuse_a, reuse_b)
  return __builtin_amdgcn_wmma_f32_16x16x32_bf16(false, a, false, b, (short)0, c,
                                                 false, false);
}

// gfx1250 async global->LDS copy (per lane: 16B), tracked by ASYNCcnt.
// Generic LDS pointer low 32 bits == wave-relative LDS byte address.
__device__ inline void lds_async_b128(void* lds_dst, const void* gsrc) {
  unsigned l = (unsigned)(size_t)lds_dst;
  asm volatile("global_load_async_to_lds_b128 %0, %1, off"
               :: "v"(l), "v"(gsrc) : "memory");
}
__device__ inline void wait_async() {
  asm volatile("s_wait_asynccnt 0x0" ::: "memory");
}

// A fragment (16x32 bf16, row-major LDS panel, stride in elements).
// ISA layout: lane<16 -> M=lane, K in [8h..8h+8) U [16+8h..16+8h+8), h=lane>>4.
__device__ inline v16bf load_a_frag(const bf16_t* Ap, int stride, int rowofs, int lane) {
  const int h = lane >> 4;
  const int r = (lane & 15) + rowofs;
  const bf16_t* p = Ap + r * stride + h * 8;
  v8bf lo = *(const v8bf*)(p);
  v8bf hi = *(const v8bf*)(p + 16);
  v16bf o;
#pragma unroll
  for (int i = 0; i < 8; ++i) { o[i] = lo[i]; o[8 + i] = hi[i]; }
  return o;
}

// B fragment: tile pre-packed so each lane's 16 bf16 are contiguous (32B/lane).
__device__ inline v16bf load_b_frag(const bf16_t* Bt, int lane) {
  return *(const v16bf*)(Bt + lane * 16);
}

// Scatter one (kk, c) weight element into WMMA B-fragment order.
__device__ inline void pack_store(bf16_t* dst, int ntc, int kk, int c, float v) {
  int tk = kk >> 5, kr = kk & 31;
  int tc = c >> 4, cr = c & 15;
  int lane = (((kr >> 3) & 1) << 4) | cr;
  int t = (kr & 7) | ((kr >> 4) << 3);
  dst[((size_t)(tk * ntc + tc) * 32 + lane) * 16 + t] = f2bf(v);
}

// ---------------- weight pack kernels (scales folded in) ----------------

__global__ __launch_bounds__(256) void k_pack_so(const float* __restrict__ w00,
                                                 const float* __restrict__ w11,
                                                 bf16_t* __restrict__ B, float c0, float c1) {
  int idx = blockIdx.x * 256 + threadIdx.x;
  if (idx >= KSO * H0c) return;
  int kk = idx >> 7, c = idx & 127;
  float v;
  if (kk < 4096) { int i = kk & 63, j = kk >> 6; v = w00[((size_t)i * 64 + j) * 128 + c] * c0; }
  else { int kp = kk - 4096, i = kp & 31, j = kp >> 5; v = w11[((size_t)i * 32 + j) * 128 + c] * c1; }
  pack_store(B, 8, kk, c, v);
}

__global__ __launch_bounds__(256) void k_pack_vo(const float* __restrict__ w01,
                                                 const float* __restrict__ w10,
                                                 bf16_t* __restrict__ B, float c0, float c1) {
  int idx = blockIdx.x * 256 + threadIdx.x;
  if (idx >= KVO * H1c) return;
  int kk = idx >> 6, c = idx & 63;
  float v;
  if (kk < 2048) { int i = kk & 63, j = kk >> 6; v = w01[((size_t)i * 32 + j) * 64 + c] * c0; }
  else { int kp = kk - 2048, i = kp & 31, j = kp >> 5; v = w10[((size_t)i * 64 + j) * 64 + c] * c1; }
  pack_store(B, 4, kk, c, v);
}

__global__ __launch_bounds__(256) void k_pack_sc(const float* __restrict__ w00,
                                                 const float* __restrict__ w11,
                                                 bf16_t* __restrict__ B, float c0, float c1) {
  int idx = blockIdx.x * 256 + threadIdx.x;
  if (idx >= KSC * A0c) return;
  int kk = idx >> 6, c = idx & 63;
  float v;
  if (kk < 8192) { int i = kk & 63, j = kk >> 6; v = w00[((size_t)i * 128 + j) * 64 + c] * c0; }
  else { int kp = kk - 8192, i = kp & 31, j = kp >> 5; v = w11[((size_t)i * 64 + j) * 64 + c] * c1; }
  pack_store(B, 4, kk, c, v);
}

__global__ __launch_bounds__(256) void k_pack_vc(const float* __restrict__ w01,
                                                 const float* __restrict__ w10,
                                                 bf16_t* __restrict__ B, float c0, float c1) {
  int idx = blockIdx.x * 256 + threadIdx.x;
  if (idx >= KVC * A1c) return;
  int kk = idx >> 5, c = idx & 31;
  float v;
  if (kk < 4096) { int i = kk & 63, j = kk >> 6; v = w01[((size_t)i * 64 + j) * 32 + c] * c0; }
  else { int kp = kk - 4096, i = kp & 31, j = kp >> 5; v = w10[((size_t)i * 128 + j) * 32 + c] * c1; }
  pack_store(B, 2, kk, c, v);
}

__global__ __launch_bounds__(256) void k_pack_mid(const float* __restrict__ w,
                                                  bf16_t* __restrict__ B, int K, float s) {
  int idx = blockIdx.x * 256 + threadIdx.x;
  if (idx >= K * K) return;
  int kk = idx / K, c = idx - kk * K;
  pack_store(B, K >> 4, kk, c, w[(size_t)c * K + kk] * s);   // B[i,k] = W[k,i]*s
}

// ---------------- open tensor product ----------------

// scalar path: 16 nodes/WG, 8 waves = 8 col tiles, K-chunk 64, dual accumulators.
__global__ __launch_bounds__(256) void k_tp_s_open(const float* __restrict__ xn,
                                                   const float* __restrict__ xe,
                                                   const bf16_t* __restrict__ B,
                                                   float* __restrict__ s0) {
  __shared__ __attribute__((aligned(16))) float xnL[16 * XW];
  __shared__ __attribute__((aligned(16))) float xeL[16 * XW];
  __shared__ __attribute__((aligned(16))) bf16_t Apan[16 * 72];
  const int tid = threadIdx.x, lane = tid & 31, wave = tid >> 5;
  const int nb = blockIdx.x * 16;
  for (int off = tid * 16; off < 16 * XW * 4; off += 256 * 16) {
    lds_async_b128((char*)xnL + off, (const char*)(xn + (size_t)nb * XW) + off);
    lds_async_b128((char*)xeL + off, (const char*)(xe + (size_t)nb * XW) + off);
  }
  wait_async();
  v8f acc0 = zero8(), acc1 = zero8();
  const int NCH = KSO / 64;  // 80 (region1: ch<64, region2: ch>=64)
  for (int ch = 0; ch < NCH; ++ch) {
    __syncthreads();
#pragma unroll
    for (int it = 0; it < 2; ++it) {
      int p = tid + it * 256;        // pair id, 512 pairs (16 rows x 32)
      int r = p >> 5;
      int kk = (p & 31) << 1;
      if (ch < 64) {                 // sn_i * se_j, j == ch
        float b = xeL[r * XW + ch];
        store_pair(&Apan[r * 72 + kk],
                   xnL[r * XW + kk] * b, xnL[r * XW + kk + 1] * b);
      } else {                       // vn_i . ve_j (3-dot)
        int i = kk & 31;
        int j = ((ch - 64) << 1) | (kk >> 5);
        const float* a = &xnL[r * XW + 64 + i * 3];
        const float* b = &xeL[r * XW + 64 + j * 3];
        store_pair(&Apan[r * 72 + kk],
                   a[0] * b[0] + a[1] * b[1] + a[2] * b[2],
                   a[3] * b[0] + a[4] * b[1] + a[5] * b[2]);
      }
    }
    __syncthreads();
    if (ch + 4 < NCH)
      __builtin_prefetch(B + ((size_t)(2 * (ch + 4)) * 8 + wave) * 512, 0, 1);
    v16bf b0 = load_b_frag(B + ((size_t)(2 * ch) * 8 + wave) * 512, lane);
    v16bf b1 = load_b_frag(B + ((size_t)(2 * ch + 1) * 8 + wave) * 512, lane);
    acc0 = wmma_bf16(load_a_frag(Apan, 72, 0, lane), b0, acc0);
    acc1 = wmma_bf16(load_a_frag(Apan + 32, 72, 0, lane), b1, acc1);
  }
  const int h = lane >> 4, c = (lane & 15) + wave * 16;
#pragma unroll
  for (int a = 0; a < 8; ++a)
    s0[(size_t)(nb + a + 8 * h) * H0c + c] = acc0[a] + acc1[a];
}

// vector path, rows (n,m): 16 nodes = 48 rows/WG, 4 waves = 4 col tiles.
__global__ __launch_bounds__(128) void k_tp_v_open(const float* __restrict__ xn,
                                                   const float* __restrict__ xe,
                                                   const bf16_t* __restrict__ B,
                                                   float* __restrict__ v0) {
  __shared__ __attribute__((aligned(16))) float xnL[16 * XW];
  __shared__ __attribute__((aligned(16))) float xeL[16 * XW];
  __shared__ __attribute__((aligned(16))) bf16_t Apan[48 * 72];
  const int tid = threadIdx.x, lane = tid & 31, wave = tid >> 5;
  const int nb = blockIdx.x * 16;
  for (int off = tid * 16; off < 16 * XW * 4; off += 128 * 16) {
    lds_async_b128((char*)xnL + off, (const char*)(xn + (size_t)nb * XW) + off);
    lds_async_b128((char*)xeL + off, (const char*)(xe + (size_t)nb * XW) + off);
  }
  wait_async();
  v8f acc[3][2];
#pragma unroll
  for (int rt = 0; rt < 3; ++rt) { acc[rt][0] = zero8(); acc[rt][1] = zero8(); }
  const int NCH = KVO / 64;  // 64 (region1: ch<32, region2: ch>=32)
  for (int ch = 0; ch < NCH; ++ch) {
    __syncthreads();
#pragma unroll
    for (int it = 0; it < 12; ++it) {
      int p = tid + it * 128;        // 1536 pairs (48 rows x 32)
      int r = p >> 5;
      int kk = (p & 31) << 1;
      int node = r / 3, m = r - node * 3;
      float v0e, v1e;
      if (ch < 32) {                 // sn_i * ve_{j,m}, j == ch
        float b = xeL[node * XW + 64 + ch * 3 + m];
        v0e = xnL[node * XW + kk] * b;
        v1e = xnL[node * XW + kk + 1] * b;
      } else {                       // vn_{i,m} * se_j
        int i = kk & 31;
        int j = ((ch - 32) << 1) | (kk >> 5);
        float b = xeL[node * XW + j];
        v0e = xnL[node * XW + 64 + i * 3 + m] * b;
        v1e = xnL[node * XW + 64 + (i + 1) * 3 + m] * b;
      }
      store_pair(&Apan[r * 72 + kk], v0e, v1e);
    }
    __syncthreads();
    if (ch + 4 < NCH)
      __builtin_prefetch(B + ((size_t)(2 * (ch + 4)) * 4 + wave) * 512, 0, 1);
    v16bf b0 = load_b_frag(B + ((size_t)(2 * ch) * 4 + wave) * 512, lane);
    v16bf b1 = load_b_frag(B + ((size_t)(2 * ch + 1) * 4 + wave) * 512, lane);
#pragma unroll
    for (int rt = 0; rt < 3; ++rt) {
      acc[rt][0] = wmma_bf16(load_a_frag(Apan, 72, rt * 16, lane), b0, acc[rt][0]);
      acc[rt][1] = wmma_bf16(load_a_frag(Apan + 32, 72, rt * 16, lane), b1, acc[rt][1]);
    }
  }
  const int h = lane >> 4, c = (lane & 15) + wave * 16;
  const int rb = nb * 3;
#pragma unroll
  for (int rt = 0; rt < 3; ++rt)
#pragma unroll
    for (int a = 0; a < 8; ++a)
      v0[(size_t)(rb + rt * 16 + a + 8 * h) * H1c + c] = acc[rt][0][a] + acc[rt][1][a];
}

// ---------------- middle layers: self-interaction + gated norm ----------------

__global__ __launch_bounds__(256) void k_mid_s(const float* __restrict__ sin_,
                                               const bf16_t* __restrict__ B,
                                               float* __restrict__ sout) {
  __shared__ __attribute__((aligned(16))) bf16_t Apan[16 * 136];
  const int tid = threadIdx.x, lane = tid & 31, wave = tid >> 5;
  const int nb = blockIdx.x * 16;
  for (int e = tid; e < 16 * H0c; e += 256) {
    int r = e >> 7, k = e & 127;
    Apan[r * 136 + k] = f2bf(sin_[(size_t)(nb + r) * H0c + k]);
  }
  __syncthreads();
  v8f acc0 = zero8(), acc1 = zero8();
#pragma unroll
  for (int ch = 0; ch < 2; ++ch) {
    v16bf a0 = load_a_frag(Apan + (2 * ch) * 32, 136, 0, lane);
    v16bf a1 = load_a_frag(Apan + (2 * ch + 1) * 32, 136, 0, lane);
    v16bf b0 = load_b_frag(B + ((size_t)(2 * ch) * 8 + wave) * 512, lane);
    v16bf b1 = load_b_frag(B + ((size_t)(2 * ch + 1) * 8 + wave) * 512, lane);
    acc0 = wmma_bf16(a0, b0, acc0);
    acc1 = wmma_bf16(a1, b1, acc1);
  }
  const int h = lane >> 4, c = (lane & 15) + wave * 16;
#pragma unroll
  for (int a = 0; a < 8; ++a) {
    float x = acc0[a] + acc1[a];
    float ns = sqrtf(x * x + EPSf);
    float g = 1.0f / ((1.0f + __expf(-ns)) * ns);   // sigmoid(ns)/ns
    sout[(size_t)(nb + a + 8 * h) * H0c + c] = x * g;
  }
}

__global__ __launch_bounds__(128) void k_mid_v(const float* __restrict__ vin,
                                               const bf16_t* __restrict__ B,
                                               float* __restrict__ vout) {
  __shared__ __attribute__((aligned(16))) bf16_t Apan[48 * 72];
  __shared__ float vT[48 * 64];
  const int tid = threadIdx.x, lane = tid & 31, wave = tid >> 5;
  const int rb = blockIdx.x * 48;
  for (int e = tid; e < 48 * H1c; e += 128) {
    int r = e >> 6, k = e & 63;
    Apan[r * 72 + k] = f2bf(vin[(size_t)(rb + r) * H1c + k]);
  }
  __syncthreads();
  v8f acc[3][2];
#pragma unroll
  for (int rt = 0; rt < 3; ++rt) { acc[rt][0] = zero8(); acc[rt][1] = zero8(); }
  v16bf b0 = load_b_frag(B + ((size_t)0 * 4 + wave) * 512, lane);
  v16bf b1 = load_b_frag(B + ((size_t)1 * 4 + wave) * 512, lane);
#pragma unroll
  for (int rt = 0; rt < 3; ++rt) {
    acc[rt][0] = wmma_bf16(load_a_frag(Apan, 72, rt * 16, lane), b0, acc[rt][0]);
    acc[rt][1] = wmma_bf16(load_a_frag(Apan + 32, 72, rt * 16, lane), b1, acc[rt][1]);
  }
  const int h = lane >> 4, c = (lane & 15) + wave * 16;
#pragma unroll
  for (int rt = 0; rt < 3; ++rt)
#pragma unroll
    for (int a = 0; a < 8; ++a)
      vT[(rt * 16 + a + 8 * h) * 64 + c] = acc[rt][0][a] + acc[rt][1][a];
  __syncthreads();
  for (int e = tid; e < 16 * 64; e += 128) {
    int node = e >> 6, k = e & 63;
    float a0 = vT[(node * 3 + 0) * 64 + k];
    float a1 = vT[(node * 3 + 1) * 64 + k];
    float a2 = vT[(node * 3 + 2) * 64 + k];
    float nv = sqrtf(a0 * a0 + a1 * a1 + a2 * a2 + EPSf);
    float g = 1.0f / ((1.0f + __expf(-nv)) * nv);
    vout[(size_t)(rb + node * 3 + 0) * H1c + k] = a0 * g;
    vout[(size_t)(rb + node * 3 + 1) * H1c + k] = a1 * g;
    vout[(size_t)(rb + node * 3 + 2) * H1c + k] = a2 * g;
  }
}

// ---------------- close tensor product ----------------

__global__ __launch_bounds__(128) void k_tp_s_close(const float* __restrict__ xn,
                                                    const float* __restrict__ s3,
                                                    const float* __restrict__ v3,
                                                    const bf16_t* __restrict__ B,
                                                    float* __restrict__ out) {
  __shared__ __attribute__((aligned(16))) float xnL[16 * XW];
  __shared__ __attribute__((aligned(16))) float sL[16 * H0c];
  __shared__ __attribute__((aligned(16))) float vL[48 * H1c];
  __shared__ __attribute__((aligned(16))) bf16_t Apan[16 * 72];
  const int tid = threadIdx.x, lane = tid & 31, wave = tid >> 5;
  const int nb = blockIdx.x * 16;
  for (int off = tid * 16; off < 16 * XW * 4; off += 128 * 16)
    lds_async_b128((char*)xnL + off, (const char*)(xn + (size_t)nb * XW) + off);
  for (int off = tid * 16; off < 16 * H0c * 4; off += 128 * 16)
    lds_async_b128((char*)sL + off, (const char*)(s3 + (size_t)nb * H0c) + off);
  for (int off = tid * 16; off < 48 * H1c * 4; off += 128 * 16)
    lds_async_b128((char*)vL + off, (const char*)(v3 + (size_t)nb * 3 * H1c) + off);
  wait_async();
  v8f acc0 = zero8(), acc1 = zero8();
  const int NCH = KSC / 64;  // 160 (region1: ch<128, region2: ch>=128)
  for (int ch = 0; ch < NCH; ++ch) {
    __syncthreads();
#pragma unroll
    for (int it = 0; it < 4; ++it) {
      int p = tid + it * 128;        // 512 pairs
      int r = p >> 5;
      int kk = (p & 31) << 1;
      if (ch < 128) {                // sn_i * s_j, j == ch
        float b = sL[r * H0c + ch];
        store_pair(&Apan[r * 72 + kk],
                   xnL[r * XW + kk] * b, xnL[r * XW + kk + 1] * b);
      } else {                       // vn_i . v_j (3-dot)
        int i = kk & 31;
        int j = ((ch - 128) << 1) | (kk >> 5);
        const float* a = &xnL[r * XW + 64 + i * 3];
        float b0v = vL[(r * 3 + 0) * H1c + j];
        float b1v = vL[(r * 3 + 1) * H1c + j];
        float b2v = vL[(r * 3 + 2) * H1c + j];
        store_pair(&Apan[r * 72 + kk],
                   a[0] * b0v + a[1] * b1v + a[2] * b2v,
                   a[3] * b0v + a[4] * b1v + a[5] * b2v);
      }
    }
    __syncthreads();
    if (ch + 4 < NCH)
      __builtin_prefetch(B + ((size_t)(2 * (ch + 4)) * 4 + wave) * 512, 0, 1);
    v16bf b0 = load_b_frag(B + ((size_t)(2 * ch) * 4 + wave) * 512, lane);
    v16bf b1 = load_b_frag(B + ((size_t)(2 * ch + 1) * 4 + wave) * 512, lane);
    acc0 = wmma_bf16(load_a_frag(Apan, 72, 0, lane), b0, acc0);
    acc1 = wmma_bf16(load_a_frag(Apan + 32, 72, 0, lane), b1, acc1);
  }
  const int h = lane >> 4, c = (lane & 15) + wave * 16;  // c < 64
#pragma unroll
  for (int a = 0; a < 8; ++a)
    out[(size_t)(nb + a + 8 * h) * XW + c] = acc0[a] + acc1[a];
}

__global__ __launch_bounds__(64) void k_tp_v_close(const float* __restrict__ xn,
                                                   const float* __restrict__ s3,
                                                   const float* __restrict__ v3,
                                                   const bf16_t* __restrict__ B,
                                                   float* __restrict__ out) {
  __shared__ __attribute__((aligned(16))) float xnL[16 * XW];
  __shared__ __attribute__((aligned(16))) float sL[16 * H0c];
  __shared__ __attribute__((aligned(16))) float vL[48 * H1c];
  __shared__ __attribute__((aligned(16))) bf16_t Apan[48 * 72];
  const int tid = threadIdx.x, lane = tid & 31, wave = tid >> 5;  // 2 col tiles
  const int nb = blockIdx.x * 16;
  for (int off = tid * 16; off < 16 * XW * 4; off += 64 * 16)
    lds_async_b128((char*)xnL + off, (const char*)(xn + (size_t)nb * XW) + off);
  for (int off = tid * 16; off < 16 * H0c * 4; off += 64 * 16)
    lds_async_b128((char*)sL + off, (const char*)(s3 + (size_t)nb * H0c) + off);
  for (int off = tid * 16; off < 48 * H1c * 4; off += 64 * 16)
    lds_async_b128((char*)vL + off, (const char*)(v3 + (size_t)nb * 3 * H1c) + off);
  wait_async();
  v8f acc[3][2];
#pragma unroll
  for (int rt = 0; rt < 3; ++rt) { acc[rt][0] = zero8(); acc[rt][1] = zero8(); }
  const int NCH = KVC / 64;  // 128 (region1: ch<64, region2: ch>=64)
  for (int ch = 0; ch < NCH; ++ch) {
    __syncthreads();
#pragma unroll
    for (int it = 0; it < 24; ++it) {
      int p = tid + it * 64;         // 1536 pairs
      int r = p >> 5;
      int kk = (p & 31) << 1;
      int node = r / 3, m = r - node * 3;
      float v0e, v1e;
      if (ch < 64) {                 // sn_i * v_{j,m}, j == ch
        float b = vL[(node * 3 + m) * H1c + ch];
        v0e = xnL[node * XW + kk] * b;
        v1e = xnL[node * XW + kk + 1] * b;
      } else {                       // vn_{i,m} * s_j
        int i = kk & 31;
        int j = ((ch - 64) << 1) | (kk >> 5);
        float b = sL[node * H0c + j];
        v0e = xnL[node * XW + 64 + i * 3 + m] * b;
        v1e = xnL[node * XW + 64 + (i + 1) * 3 + m] * b;
      }
      store_pair(&Apan[r * 72 + kk], v0e, v1e);
    }
    __syncthreads();
    if (ch + 4 < NCH)
      __builtin_prefetch(B + ((size_t)(2 * (ch + 4)) * 2 + wave) * 512, 0, 1);
    v16bf b0 = load_b_frag(B + ((size_t)(2 * ch) * 2 + wave) * 512, lane);
    v16bf b1 = load_b_frag(B + ((size_t)(2 * ch + 1) * 2 + wave) * 512, lane);
#pragma unroll
    for (int rt = 0; rt < 3; ++rt) {
      acc[rt][0] = wmma_bf16(load_a_frag(Apan, 72, rt * 16, lane), b0, acc[rt][0]);
      acc[rt][1] = wmma_bf16(load_a_frag(Apan + 32, 72, rt * 16, lane), b1, acc[rt][1]);
    }
  }
  const int h = lane >> 4, k = (lane & 15) + wave * 16;  // k < 32
#pragma unroll
  for (int rt = 0; rt < 3; ++rt) {
#pragma unroll
    for (int a = 0; a < 8; ++a) {
      int r = rt * 16 + a + 8 * h;
      int node = r / 3, m = r - node * 3;
      out[(size_t)(nb + node) * XW + 64 + k * 3 + m] = acc[rt][0][a] + acc[rt][1][a];
    }
  }
}

// ---------------- host launch ----------------

extern "C" void kernel_launch(void* const* d_in, const int* in_sizes, int n_in,
                              void* d_out, int out_size, void* d_ws, size_t ws_size,
                              hipStream_t stream) {
  (void)in_sizes; (void)n_in; (void)out_size;
  const float* xn   = (const float*)d_in[0];
  const float* xe   = (const float*)d_in[1];
  const float* w00o = (const float*)d_in[2];
  const float* w01o = (const float*)d_in[3];
  const float* w10o = (const float*)d_in[4];
  const float* w11o = (const float*)d_in[5];
  const float* wsL[3] = {(const float*)d_in[6], (const float*)d_in[8], (const float*)d_in[10]};
  const float* wvL[3] = {(const float*)d_in[7], (const float*)d_in[9], (const float*)d_in[11]};
  const float* w00c = (const float*)d_in[12];
  const float* w01c = (const float*)d_in[13];
  const float* w10c = (const float*)d_in[14];
  const float* w11c = (const float*)d_in[15];
  float* out = (float*)d_out;
  if (ws_size < WS_NEED) return;  // need ~46 MB scratch

  char* wsb = (char*)d_ws;
  bf16_t* Bso = (bf16_t*)(wsb + OFF_BSO);
  bf16_t* Bvo = (bf16_t*)(wsb + OFF_BVO);
  bf16_t* Bsc = (bf16_t*)(wsb + OFF_BSC);
  bf16_t* Bvc = (bf16_t*)(wsb + OFF_BVC);
  bf16_t* Bms = (bf16_t*)(wsb + OFF_BMS);
  bf16_t* Bmv = (bf16_t*)(wsb + OFF_BMV);
  float* sA = (float*)(wsb + OFF_SA);
  float* sB = (float*)(wsb + OFF_SB);
  float* vA = (float*)(wsb + OFF_VA);
  float* vB = (float*)(wsb + OFF_VB);

  const float is2  = 0.70710678118654752f;
  const float c00o = is2 / sqrtf(64.f * 64.f);
  const float c11o = is2 / sqrtf(3.f * 32.f * 32.f);
  const float c01o = is2 / sqrtf(64.f * 32.f);
  const float c10o = is2 / sqrtf(32.f * 64.f);
  const float c00c = is2 / sqrtf(64.f * 128.f);
  const float c11c = is2 / sqrtf(3.f * 32.f * 64.f);
  const float c01c = is2 / sqrtf(64.f * 64.f);
  const float c10c = is2 / sqrtf(32.f * 128.f);
  const float cms  = 1.f / sqrtf(128.f);
  const float cmv  = 1.f / sqrtf(64.f);

  k_pack_so<<<(KSO * H0c + 255) / 256, 256, 0, stream>>>(w00o, w11o, Bso, c00o, c11o);
  k_pack_vo<<<(KVO * H1c + 255) / 256, 256, 0, stream>>>(w01o, w10o, Bvo, c01o, c10o);
  k_pack_sc<<<(KSC * A0c + 255) / 256, 256, 0, stream>>>(w00c, w11c, Bsc, c00c, c11c);
  k_pack_vc<<<(KVC * A1c + 255) / 256, 256, 0, stream>>>(w01c, w10c, Bvc, c01c, c10c);
  for (int l = 0; l < 3; ++l) {
    k_pack_mid<<<(H0c * H0c + 255) / 256, 256, 0, stream>>>(wsL[l], Bms + l * H0c * H0c, H0c, cms);
    k_pack_mid<<<(H1c * H1c + 255) / 256, 256, 0, stream>>>(wvL[l], Bmv + l * H1c * H1c, H1c, cmv);
  }

  const int NG = NN / 16;  // 1024 node tiles
  k_tp_s_open<<<NG, 256, 0, stream>>>(xn, xe, Bso, sA);
  k_tp_v_open<<<NG, 128, 0, stream>>>(xn, xe, Bvo, vA);

  float* si = sA; float* so = sB; float* vi = vA; float* vo = vB;
  for (int l = 0; l < 3; ++l) {
    k_mid_s<<<NG, 256, 0, stream>>>(si, Bms + l * H0c * H0c, so);
    k_mid_v<<<NG, 128, 0, stream>>>(vi, Bmv + l * H1c * H1c, vo);
    float* t = si; si = so; so = t;
    t = vi; vi = vo; vo = t;
  }
  k_tp_s_close<<<NG, 128, 0, stream>>>(xn, si, vi, Bsc, out);
  k_tp_v_close<<<NG, 64, 0, stream>>>(xn, si, vi, Bvc, out);
}
